// HyperbolicRecurrentRGCN_69578470195520
// MI455X (gfx1250) — compile-verified, matrix operand order checked by reading, not run on hardware
//
#include <hip/hip_runtime.h>
#include <math.h>

typedef __attribute__((ext_vector_type(16))) __bf16 v16bf;
typedef __attribute__((ext_vector_type(8)))  __bf16 v8bf;
typedef __attribute__((ext_vector_type(4)))  __bf16 v4bf;
typedef __attribute__((ext_vector_type(8)))  float  v8f;

#define N_ENT   40000
#define DDIM    128
#define T_STEPS 8
#define E_EDGES 200000
#define SQC     0.1f                    // sqrt(0.01)
#define EPS_F   1e-6f
#define RRELU_S 0.22916666666666666f    // (1/8 + 1/3)/2

// ---------- helpers ----------

__device__ __forceinline__ __bf16 ftobf(float f) {
    return static_cast<__bf16>(f);      // native v_cvt bf16 path on gfx1250
}

// K index inside a 16x32 bf16 A fragment (ISA 7.12.2, 16-bit A-matrix layout).
__device__ __forceinline__ int frag_k(int v, int lo, int khalf) {
    int base = (v < 4) ? (v * 2) : (16 + (v - 4) * 2);
    return base + lo + khalf * 8;
}

union FragU { v16bf v; v8bf h[2]; };

// Build A fragment from a bf16 row pointer: two contiguous 16B runs per lane.
__device__ __forceinline__ v16bf load_afrag(const __bf16* rowp, int kt, int khalf) {
    FragU u;
    u.h[0] = *(const v8bf*)(rowp + kt * 32 + khalf * 8);
    u.h[1] = *(const v8bf*)(rowp + kt * 32 + 16 + khalf * 8);
    return u.v;
}

__device__ __forceinline__ float wave_sum(float v) {
    #pragma unroll
    for (int m = 16; m >= 1; m >>= 1) v += __shfl_xor(v, m, 32);
    return v;
}

// ---------- weight pre-swizzle into WMMA B-fragment order ----------
// Layout: [w][ntile(8)][ktile(4)][lane(32)][elem(16)]  (bf16)
__global__ void wconv_kernel(const float* __restrict__ w0,
                             const float* __restrict__ w1,
                             const float* __restrict__ w2,
                             __bf16* __restrict__ wsw) {
    int tid = blockIdx.x * blockDim.x + threadIdx.x;
    if (tid >= 3 * DDIM * DDIM) return;
    int w = tid / (DDIM * DDIM);
    int p = tid - w * (DDIM * DDIM);
    int elem  = p & 15;
    int lane  = (p >> 4) & 31;
    int ktile = (p >> 9) & 3;
    int ntile = (p >> 11) & 7;
    int khalf = lane >> 4, nn = lane & 15;
    int k = ktile * 32 + frag_k(elem >> 1, elem & 1, khalf);
    int n = ntile * 16 + nn;
    const float* src = (w == 0) ? w0 : (w == 1) ? w1 : w2;
    wsw[tid] = ftobf(src[k * DDIM + n]);
}

// ---------- logmap0 per entity row (wave per row) ----------
__global__ __launch_bounds__(128) void logmap_kernel(const float* __restrict__ h,
                                                     float* __restrict__ htan,
                                                     __bf16* __restrict__ htan_bf) {
    int row  = blockIdx.x * 4 + (threadIdx.x >> 5);
    int lane = threadIdx.x & 31;
    if (row >= N_ENT) return;
    float4 u = ((const float4*)(h + (size_t)row * DDIM))[lane];
    float ss = wave_sum(u.x * u.x + u.y * u.y + u.z * u.z + u.w * u.w);
    float n = fmaxf(sqrtf(ss), EPS_F);
    float t = fminf(SQC * n, 1.0f - 1e-5f);
    float s = atanhf(t) / (SQC * n);
    float4 o = make_float4(u.x * s, u.y * s, u.z * s, u.w * s);
    ((float4*)(htan + (size_t)row * DDIM))[lane] = o;
    v4bf b;
    b[0] = ftobf(o.x); b[1] = ftobf(o.y); b[2] = ftobf(o.z); b[3] = ftobf(o.w);
    ((v4bf*)(htan_bf + (size_t)row * DDIM))[lane] = b;
}

// ---------- zero agg + cnt ----------
__global__ void zero_kernel(float* __restrict__ agg, float* __restrict__ cnt) {
    int tid = blockIdx.x * blockDim.x + threadIdx.x;
    if (tid < N_ENT * DDIM) agg[tid] = 0.0f;
    if (tid < N_ENT)        cnt[tid] = 0.0f;
}

// ---------- per-destination edge counts ----------
__global__ void cnt_kernel(const int* __restrict__ dst, float* __restrict__ cnt) {
    int e = blockIdx.x * blockDim.x + threadIdx.x;
    if (e < E_EDGES) atomicAdd(&cnt[dst[e]], 1.0f);
}

// ---------- edge message GEMM + fused scatter-add ----------
// One wave handles 16 edges: msg[16x128] = (htan[src]+emb_rel[rel]) @ w_neigh,
// then atomicAdd into agg[dst]. Gathered rows staged in LDS as bf16 row-major
// so A fragments are two ds_load_b128 per K-tile.
__global__ __launch_bounds__(128) void edge_msg_kernel(
        const int* __restrict__ src, const int* __restrict__ dst,
        const int* __restrict__ rel,
        const float* __restrict__ htan, const float* __restrict__ emb_rel,
        const __bf16* __restrict__ wn, float* __restrict__ agg) {
    __shared__ __bf16 smem[4][16][DDIM];   // 16 KB
    int wave = threadIdx.x >> 5;
    int lane = threadIdx.x & 31;
    int base = (blockIdx.x * 4 + wave) * 16;

    // Stage row j (wave-uniform indices, coalesced float4 gather, packed cvt).
    #pragma unroll
    for (int j = 0; j < 16; ++j) {
        int s = src[base + j];
        int r = rel[base + j];
        float4 a = ((const float4*)(htan    + (size_t)s * DDIM))[lane];
        float4 b = ((const float4*)(emb_rel + (size_t)r * DDIM))[lane];
        v4bf o;
        o[0] = ftobf(a.x + b.x); o[1] = ftobf(a.y + b.y);
        o[2] = ftobf(a.z + b.z); o[3] = ftobf(a.w + b.w);
        ((v4bf*)&smem[wave][j][0])[lane] = o;
    }
    __syncthreads();

    int m = lane & 15, khalf = lane >> 4;
    const __bf16* rowp = &smem[wave][m][0];
    v16bf afrag[4];
    #pragma unroll
    for (int kt = 0; kt < 4; ++kt) afrag[kt] = load_afrag(rowp, kt, khalf);

    v8f zero8 = {0.f, 0.f, 0.f, 0.f, 0.f, 0.f, 0.f, 0.f};
    v8f acc[8];
    #pragma unroll
    for (int nt = 0; nt < 8; ++nt) acc[nt] = zero8;

    #pragma unroll
    for (int nt = 0; nt < 8; ++nt)
        #pragma unroll
        for (int kt = 0; kt < 4; ++kt) {
            v16bf bfrag = *(const v16bf*)(wn + ((nt * 4 + kt) * 32 + lane) * 16);
            acc[nt] = __builtin_amdgcn_wmma_f32_16x16x32_bf16(
                false, afrag[kt], false, bfrag, (short)0, acc[nt], false, false);
        }

    // C layout: VGPR v of lane holds row (v + 8*khalf), col (nt*16 + lane&15)
    #pragma unroll
    for (int v = 0; v < 8; ++v) {
        int e    = base + v + 8 * khalf;
        int drow = dst[e];
        float* outp = agg + (size_t)drow * DDIM + (lane & 15);
        #pragma unroll
        for (int nt = 0; nt < 8; ++nt)
            atomicAdd(outp + nt * 16, acc[nt][v]);
    }
}

// ---------- self-loop + gate GEMMs, mean, rrelu ----------
// One wave: 16 entity rows; out_tan written in place over agg, gate_lin to glin.
__global__ __launch_bounds__(128) void loop_gate_kernel(
        const __bf16* __restrict__ htan_bf,
        const __bf16* __restrict__ wl, const __bf16* __restrict__ wg,
        float* __restrict__ agg, const float* __restrict__ cnt,
        float* __restrict__ glin) {
    int lane = threadIdx.x & 31;
    int base = (blockIdx.x * 4 + (threadIdx.x >> 5)) * 16;
    int m = lane & 15, khalf = lane >> 4;

    const __bf16* rowp = htan_bf + (size_t)(base + m) * DDIM;
    v16bf afrag[4];
    #pragma unroll
    for (int kt = 0; kt < 4; ++kt) afrag[kt] = load_afrag(rowp, kt, khalf);

    v8f zero8 = {0.f, 0.f, 0.f, 0.f, 0.f, 0.f, 0.f, 0.f};
    v8f accL[8], accG[8];
    #pragma unroll
    for (int nt = 0; nt < 8; ++nt) { accL[nt] = zero8; accG[nt] = zero8; }

    #pragma unroll
    for (int nt = 0; nt < 8; ++nt)
        #pragma unroll
        for (int kt = 0; kt < 4; ++kt) {
            v16bf bl = *(const v16bf*)(wl + ((nt * 4 + kt) * 32 + lane) * 16);
            v16bf bg = *(const v16bf*)(wg + ((nt * 4 + kt) * 32 + lane) * 16);
            accL[nt] = __builtin_amdgcn_wmma_f32_16x16x32_bf16(
                false, afrag[kt], false, bl, (short)0, accL[nt], false, false);
            accG[nt] = __builtin_amdgcn_wmma_f32_16x16x32_bf16(
                false, afrag[kt], false, bg, (short)0, accG[nt], false, false);
        }

    #pragma unroll
    for (int v = 0; v < 8; ++v) {
        int row = base + v + 8 * khalf;
        float inv = 1.0f / fmaxf(cnt[row], 1.0f);
        #pragma unroll
        for (int nt = 0; nt < 8; ++nt) {
            int idx = row * DDIM + nt * 16 + (lane & 15);
            float lv = accL[nt][v] + agg[idx] * inv;     // agg mean + self loop
            lv = (lv >= 0.0f) ? lv : RRELU_S * lv;       // rrelu (eval)
            agg[idx]  = lv;                              // out_tan (in place)
            glin[idx] = accG[nt][v];                     // gate pre-activation
        }
    }
}

// ---------- hyperbolic maps + gate + final expmap0 (wave per row) ----------
__global__ __launch_bounds__(128) void post_kernel(
        const float* __restrict__ outtan, const float* __restrict__ glin,
        const float* __restrict__ htan, float* __restrict__ hout) {
    int row  = blockIdx.x * 4 + (threadIdx.x >> 5);
    int lane = threadIdx.x & 31;
    if (row >= N_ENT) return;

    float4 u = ((const float4*)(outtan + (size_t)row * DDIM))[lane];
    float ss = wave_sum(u.x * u.x + u.y * u.y + u.z * u.z + u.w * u.w);
    float n1 = fmaxf(sqrtf(ss), EPS_F);
    float th = tanhf(SQC * n1);
    float s1 = th / (SQC * n1);                // expmap0 scale
    float n2 = fmaxf(th / SQC, EPS_F);         // ||expmap0(u)|| (scalar)
    float t2 = fminf(SQC * n2, 1.0f - 1e-5f);
    float s2 = atanhf(t2) / (SQC * n2);        // logmap0 scale
    float sc = s1 * s2;                        // hnew_tan = sc*u

    float4 g4 = ((const float4*)(glin + (size_t)row * DDIM))[lane];
    float4 ht = ((const float4*)(htan + (size_t)row * DDIM))[lane];
    float gx = 1.0f / (1.0f + expf(-g4.x));
    float gy = 1.0f / (1.0f + expf(-g4.y));
    float gz = 1.0f / (1.0f + expf(-g4.z));
    float gw = 1.0f / (1.0f + expf(-g4.w));
    float cx = gx * (sc * u.x) + (1.0f - gx) * ht.x;
    float cy = gy * (sc * u.y) + (1.0f - gy) * ht.y;
    float cz = gz * (sc * u.z) + (1.0f - gz) * ht.z;
    float cw = gw * (sc * u.w) + (1.0f - gw) * ht.w;

    float ss3 = wave_sum(cx * cx + cy * cy + cz * cz + cw * cw);
    float n3 = fmaxf(sqrtf(ss3), EPS_F);
    float s3 = tanhf(SQC * n3) / (SQC * n3);   // final expmap0
    ((float4*)(hout + (size_t)row * DDIM))[lane] =
        make_float4(cx * s3, cy * s3, cz * s3, cw * s3);
}

// ---------- driver ----------
extern "C" void kernel_launch(void* const* d_in, const int* in_sizes, int n_in,
                              void* d_out, int out_size, void* d_ws, size_t ws_size,
                              hipStream_t stream) {
    const int*   src     = (const int*)d_in[0];
    const int*   dst     = (const int*)d_in[1];
    const int*   rel     = (const int*)d_in[2];
    const float* dyn     = (const float*)d_in[3];
    const float* emb_rel = (const float*)d_in[4];
    const float* w_neigh = (const float*)d_in[5];
    const float* w_loop  = (const float*)d_in[6];
    const float* w_gate  = (const float*)d_in[7];
    float* out = (float*)d_out;

    char* ws = (char*)d_ws;
    size_t off = 0;
    auto carve = [&](size_t bytes) -> void* {
        void* p = ws + off;
        off += (bytes + 255) & ~(size_t)255;
        return p;
    };
    float*  h       = (float*)carve((size_t)N_ENT * DDIM * sizeof(float));
    float*  htan    = (float*)carve((size_t)N_ENT * DDIM * sizeof(float));
    __bf16* htan_bf = (__bf16*)carve((size_t)N_ENT * DDIM * sizeof(__bf16));
    float*  agg     = (float*)carve((size_t)N_ENT * DDIM * sizeof(float));
    float*  glin    = (float*)carve((size_t)N_ENT * DDIM * sizeof(float));
    float*  cnt     = (float*)carve((size_t)N_ENT * sizeof(float));
    __bf16* wsw     = (__bf16*)carve((size_t)3 * DDIM * DDIM * sizeof(__bf16));
    (void)ws_size; (void)in_sizes; (void)n_in; (void)out_size;

    wconv_kernel<<<(3 * DDIM * DDIM + 255) / 256, 256, 0, stream>>>(
        w_neigh, w_loop, w_gate, wsw);

    for (int t = 0; t < T_STEPS; ++t) {
        const float* hin = (t == 0) ? dyn : h;
        logmap_kernel<<<N_ENT / 4, 128, 0, stream>>>(hin, htan, htan_bf);
        zero_kernel<<<(N_ENT * DDIM + 255) / 256, 256, 0, stream>>>(agg, cnt);
        edge_msg_kernel<<<E_EDGES / 64, 128, 0, stream>>>(
            src + (size_t)t * E_EDGES, dst + (size_t)t * E_EDGES,
            rel + (size_t)t * E_EDGES, htan, emb_rel, wsw, agg);
        cnt_kernel<<<(E_EDGES + 255) / 256, 256, 0, stream>>>(
            dst + (size_t)t * E_EDGES, cnt);
        loop_gate_kernel<<<N_ENT / 64, 128, 0, stream>>>(
            htan_bf, wsw + DDIM * DDIM, wsw + 2 * DDIM * DDIM, agg, cnt, glin);
        post_kernel<<<N_ENT / 4, 128, 0, stream>>>(
            agg, glin, htan, (t == T_STEPS - 1) ? out : h);
    }
}
